// Backbone3D_46566035423478
// MI455X (gfx1250) — matrix operand order repaired and checked by reading, compiled, not run on hardware
//
#include <hip/hip_runtime.h>

typedef __bf16 bf16_t;
typedef __attribute__((ext_vector_type(16))) __bf16 v16bf;
typedef __attribute__((ext_vector_type(8)))  __bf16 v8bf;
typedef __attribute__((ext_vector_type(8)))  float  v8f;

struct ConvArgs {
  int Bn, Di, Hi, Wi, Do, Ho, Wo;
  int cin, cout;
  int kd, kh, kw, sd, sh, sw, pd, ph, pw;
  int M;         // Bn*Do*Ho*Wo
  int cpt;       // ceil(cin/32): K-chunks per tap
  int nchunks;   // taps * cpt
  int instride;  // padded channel stride of input activations (bf16 elems)
  int ostride;   // padded channel stride of output activations
  int outF32;    // 1 -> write f32 (final layer to d_out), else bf16
};

// ---------------------------------------------------------------- utilities
__global__ void zero_f32_kernel(float* __restrict__ p, size_t n) {
  size_t i = (size_t)blockIdx.x * blockDim.x + threadIdx.x;
  if (i < n) p[i] = 0.0f;
}

__global__ void scatter_kernel(const float* __restrict__ feats,
                               const int* __restrict__ coords,
                               bf16_t* __restrict__ dense, float* __restrict__ mask,
                               int nvox, int D, int H, int W, int stride) {
  int i = blockIdx.x * blockDim.x + threadIdx.x;
  if (i >= nvox) return;
  int b = coords[i * 4 + 0];
  int z = coords[i * 4 + 1];
  int y = coords[i * 4 + 2];
  int x = coords[i * 4 + 3];
  size_t s = (((size_t)b * D + z) * H + y) * W + x;
  for (int c = 0; c < 4; ++c) dense[s * stride + c] = (bf16_t)feats[i * 4 + c];
  mask[s] = 1.0f;
}

// out-mask for strided (non-subm) layers: active if any input in RF is active
__global__ void mask_down_kernel(const float* __restrict__ mIn,
                                 float* __restrict__ mOut, ConvArgs a) {
  int s = blockIdx.x * blockDim.x + threadIdx.x;
  if (s >= a.M) return;
  int xo = s % a.Wo; int t = s / a.Wo;
  int yo = t % a.Ho; t /= a.Ho;
  int zo = t % a.Do; int b = t / a.Do;
  float m = 0.0f;
  for (int tz = 0; tz < a.kd; ++tz) {
    int zi = zo * a.sd - a.pd + tz;
    if (zi < 0 || zi >= a.Di) continue;
    for (int ty = 0; ty < a.kh; ++ty) {
      int yi = yo * a.sh - a.ph + ty;
      if (yi < 0 || yi >= a.Hi) continue;
      for (int tx = 0; tx < a.kw; ++tx) {
        int xi = xo * a.sw - a.pw + tx;
        if (xi < 0 || xi >= a.Wi) continue;
        size_t off = (((size_t)b * a.Di + zi) * a.Hi + yi) * a.Wi + xi;
        if (mIn[off] > 0.0f) m = 1.0f;
      }
    }
  }
  mOut[s] = m;
}

// Repack f32 weights [taps][cin][cout] into per-lane B operand order for
// v_wmma_f32_16x16x32_bf16: element ((ntile*nchunks + chunk)*32 + lane)*16 + j
// holds B[K = (chunk%cpt)*32 + (lane>>4)*8 + (j&7) + (j>>3)*16][n = ntile*16 + (lane&15)],
// zero-padded where K >= cin.
__global__ void repack_w_kernel(const float* __restrict__ w, bf16_t* __restrict__ wp,
                                int cin, int cout, int cpt, int nchunks, int total) {
  int i = blockIdx.x * blockDim.x + threadIdx.x;
  if (i >= total) return;
  int j     = i & 15;
  int lane  = (i >> 4) & 31;
  int chunk = (i >> 9) % nchunks;
  int ntile = (i >> 9) / nchunks;
  int n   = ntile * 16 + (lane & 15);
  int tap = chunk / cpt;
  int kc  = (chunk % cpt) * 32 + (lane >> 4) * 8 + (j & 7) + (j >> 3) * 16;
  float v = 0.0f;
  if (kc < cin) v = w[((size_t)tap * cin + kc) * cout + n];
  wp[i] = (bf16_t)v;
}

// ------------------------------------------------- implicit-GEMM conv (WMMA)
// One wave per 16 output sites; NT = cout/16 accumulator tiles per wave.
// A fragments load straight from the padded bf16 activation rows (16B loads),
// B fragments from the pre-swizzled weight pack (32B contiguous per lane).
template <int NT>
__global__ void __launch_bounds__(32)
conv_wmma_kernel(const bf16_t* __restrict__ in, const bf16_t* __restrict__ wp,
                 const float* __restrict__ scale, const float* __restrict__ bias,
                 const float* __restrict__ maskOut, void* __restrict__ outPtr,
                 ConvArgs a) {
  __shared__ int   sSite[16];
  __shared__ float sMask[16];

  const int lane = threadIdx.x;
  const int half = lane >> 4;
  const int l15  = lane & 15;
  const int mtile = blockIdx.x;

  // site for row m = l15 (both halves compute identical values)
  const int sglob = mtile * 16 + l15;
  const int siteValid = (sglob < a.M) ? 1 : 0;
  const int ss = siteValid ? sglob : 0;
  int xo = ss % a.Wo; int t = ss / a.Wo;
  int yo = t % a.Ho;  t /= a.Ho;
  int zo = t % a.Do;  const int bb = t / a.Do;

  if (lane < 16) {
    sSite[l15] = siteValid ? sglob : -1;
    sMask[l15] = siteValid ? maskOut[sglob] : 0.0f;
  }
  __syncthreads();

  float anyA = 0.0f;
  for (int i = 0; i < 16; ++i) anyA += sMask[i];  // uniform across wave

  v8f acc[NT];
#pragma unroll
  for (int nt = 0; nt < NT; ++nt) acc[nt] = {};

  if (anyA != 0.0f) {
    int tap = 0;
    for (int tz = 0; tz < a.kd; ++tz)
      for (int ty = 0; ty < a.kh; ++ty)
        for (int tx = 0; tx < a.kw; ++tx, ++tap) {
          int zi = zo * a.sd - a.pd + tz;
          int yi = yo * a.sh - a.ph + ty;
          int xi = xo * a.sw - a.pw + tx;
          const int ok = siteValid && zi >= 0 && zi < a.Di && yi >= 0 &&
                         yi < a.Hi && xi >= 0 && xi < a.Wi;
          const int off =
              ((((bb * a.Di + zi) * a.Hi + yi) * a.Wi) + xi) * a.instride;

          for (int cc = 0; cc < a.cpt; ++cc) {
            // A fragment (ISA 16-bit A layout): lane<16 gets K 0-7/16-23 of
            // its row, lane>=16 gets K 8-15/24-31; rows are contiguous K.
            v16bf av;
#pragma unroll
            for (int j = 0; j < 16; ++j) av[j] = (bf16_t)0.0f;
            if (ok) {
              const bf16_t* ap = in + off + cc * 32 + half * 8;
              v8bf a0 = *(const v8bf*)(ap);
              v8bf a1 = *(const v8bf*)(ap + 16);
#pragma unroll
              for (int j = 0; j < 8; ++j) { av[j] = a0[j]; av[j + 8] = a1[j]; }
            }

            const int chunk = tap * a.cpt + cc;
#pragma unroll
            for (int nt = 0; nt < NT; ++nt) {
              const bf16_t* bp =
                  wp + ((size_t)(nt * a.nchunks + chunk) * 32 + lane) * 16;
              v8bf b0 = *(const v8bf*)(bp);
              v8bf b1 = *(const v8bf*)(bp + 8);
              v16bf bv;
#pragma unroll
              for (int j = 0; j < 8; ++j) { bv[j] = b0[j]; bv[j + 8] = b1[j]; }
              acc[nt] = __builtin_amdgcn_wmma_f32_16x16x32_bf16(
                  false, av, false, bv, (short)0, acc[nt], false, false);
            }
          }
        }
  }

  // epilogue: relu(acc*scale + bias) * mask ; D layout: VGPR r -> M = r+8*half
#pragma unroll
  for (int nt = 0; nt < NT; ++nt) {
    const int n = nt * 16 + l15;
    const float sc = scale[n];
    const float bi = bias[n];
#pragma unroll
    for (int r = 0; r < 8; ++r) {
      int m = r + half * 8;
      int s = sSite[m];
      if (s >= 0) {
        float v = acc[nt][r] * sc + bi;
        v = v > 0.0f ? v : 0.0f;
        v *= sMask[m];
        if (a.outF32)
          ((float*)outPtr)[(size_t)s * a.ostride + n] = v;
        else
          ((bf16_t*)outPtr)[(size_t)s * a.ostride + n] = (bf16_t)v;
      }
    }
  }
}

// ------------------------------------------------------------------- driver
struct LayerDef {
  int kd, kh, kw, cin, cout, sd, sh, sw, pd, ph, pw, subm;
};
static const LayerDef kLayers[14] = {
    {3, 3, 3, 4, 16, 1, 1, 1, 1, 1, 1, 1},
    {3, 3, 3, 16, 16, 1, 1, 1, 1, 1, 1, 1},
    {3, 3, 3, 16, 32, 2, 2, 2, 1, 1, 1, 0},
    {3, 3, 3, 32, 32, 1, 1, 1, 1, 1, 1, 1},
    {3, 3, 3, 32, 32, 1, 1, 1, 1, 1, 1, 1},
    {3, 3, 3, 32, 64, 2, 2, 2, 1, 1, 1, 0},
    {3, 3, 3, 64, 64, 1, 1, 1, 1, 1, 1, 1},
    {3, 3, 3, 64, 64, 1, 1, 1, 1, 1, 1, 1},
    {3, 3, 3, 64, 64, 1, 1, 1, 1, 1, 1, 1},
    {3, 3, 3, 64, 64, 2, 2, 2, 0, 1, 1, 0},
    {3, 3, 3, 64, 64, 1, 1, 1, 1, 1, 1, 1},
    {3, 3, 3, 64, 64, 1, 1, 1, 1, 1, 1, 1},
    {3, 3, 3, 64, 64, 1, 1, 1, 1, 1, 1, 1},
    {3, 1, 1, 64, 128, 2, 1, 1, 0, 0, 0, 0},
};

extern "C" void kernel_launch(void* const* d_in, const int* in_sizes, int n_in,
                              void* d_out, int out_size, void* d_ws, size_t ws_size,
                              hipStream_t stream) {
  (void)n_in; (void)out_size; (void)ws_size;
  const float* vfeat  = (const float*)d_in[0];
  const int*   vcoord = (const int*)d_in[1];
  const int nvox = in_sizes[1] / 4;

  size_t off = 0;
  auto carve = [&](size_t bytes) -> char* {
    char* p = (char*)d_ws + off;
    off += (bytes + 255) & ~(size_t)255;
    return p;
  };

  const int B0 = 2, D0 = 41, H0 = 160, W0 = 140;
  const size_t M0 = (size_t)B0 * D0 * H0 * W0;

  bf16_t* featA = (bf16_t*)carve(M0 * 32 * sizeof(bf16_t));
  bf16_t* featB = (bf16_t*)carve(M0 * 32 * sizeof(bf16_t));
  bf16_t* wpack = (bf16_t*)carve((size_t)262144 * sizeof(bf16_t));
  float*  mask0 = (float*)carve(M0 * sizeof(float));

  {
    // zero both padded bf16 activation buffers (pad channels must be 0) + mask
    size_t nzf = M0 * 32 / 2;  // bf16 elems -> f32 words
    zero_f32_kernel<<<(unsigned)((nzf + 255) / 256), 256, 0, stream>>>(
        (float*)featA, nzf);
    zero_f32_kernel<<<(unsigned)((nzf + 255) / 256), 256, 0, stream>>>(
        (float*)featB, nzf);
    zero_f32_kernel<<<(unsigned)((M0 + 255) / 256), 256, 0, stream>>>(mask0, M0);
    scatter_kernel<<<(nvox + 255) / 256, 256, 0, stream>>>(
        vfeat, vcoord, featA, mask0, nvox, D0, H0, W0, 32);
  }

  int Di = D0, Hi = H0, Wi = W0;
  bf16_t* bufs[2] = {featA, featB};
  int cur = 0;
  float* curMask = mask0;

  for (int i = 0; i < 14; ++i) {
    const LayerDef& L = kLayers[i];
    const int Do = (Di + 2 * L.pd - L.kd) / L.sd + 1;
    const int Ho = (Hi + 2 * L.ph - L.kh) / L.sh + 1;
    const int Wo = (Wi + 2 * L.pw - L.kw) / L.sw + 1;
    const int taps = L.kd * L.kh * L.kw;
    const int cpt = (L.cin + 31) / 32;
    const int nchunks = taps * cpt;
    const int NT = L.cout / 16;

    ConvArgs a;
    a.Bn = B0; a.Di = Di; a.Hi = Hi; a.Wi = Wi;
    a.Do = Do; a.Ho = Ho; a.Wo = Wo;
    a.cin = L.cin; a.cout = L.cout;
    a.kd = L.kd; a.kh = L.kh; a.kw = L.kw;
    a.sd = L.sd; a.sh = L.sh; a.sw = L.sw;
    a.pd = L.pd; a.ph = L.ph; a.pw = L.pw;
    a.M = B0 * Do * Ho * Wo;
    a.cpt = cpt; a.nchunks = nchunks;
    a.instride = (L.cin < 32) ? 32 : L.cin;
    a.ostride = (i == 13) ? L.cout : ((L.cout < 32) ? 32 : L.cout);
    a.outF32 = (i == 13) ? 1 : 0;

    // repack this layer's weights to WMMA B-operand order (bf16)
    const int wtot = NT * nchunks * 512;
    repack_w_kernel<<<(wtot + 255) / 256, 256, 0, stream>>>(
        (const float*)d_in[2 + i], wpack, L.cin, L.cout, cpt, nchunks, wtot);

    // downsample layers: compute new active-site mask
    float* outMask = curMask;
    if (!L.subm) {
      outMask = (float*)carve((size_t)a.M * sizeof(float));
      mask_down_kernel<<<(a.M + 255) / 256, 256, 0, stream>>>(curMask, outMask, a);
    }

    void* outBuf = (i == 13) ? d_out : (void*)bufs[cur ^ 1];
    dim3 grid((a.M + 15) / 16);
    const float* sc = (const float*)d_in[16 + i];
    const float* bi = (const float*)d_in[30 + i];
    switch (NT) {
      case 1:
        conv_wmma_kernel<1><<<grid, 32, 0, stream>>>(bufs[cur], wpack, sc, bi,
                                                     outMask, outBuf, a);
        break;
      case 2:
        conv_wmma_kernel<2><<<grid, 32, 0, stream>>>(bufs[cur], wpack, sc, bi,
                                                     outMask, outBuf, a);
        break;
      case 4:
        conv_wmma_kernel<4><<<grid, 32, 0, stream>>>(bufs[cur], wpack, sc, bi,
                                                     outMask, outBuf, a);
        break;
      default:
        conv_wmma_kernel<8><<<grid, 32, 0, stream>>>(bufs[cur], wpack, sc, bi,
                                                     outMask, outBuf, a);
        break;
    }

    cur ^= 1;
    curMask = outMask;
    Di = Do; Hi = Ho; Wi = Wo;
  }
}
